// OurConv1D_90039694394027
// MI455X (gfx1250) — compile-verified, matrix operand order checked by reading, not run on hardware
//
#include <hip/hip_runtime.h>
#include <hip/hip_bf16.h>
#include <cstdint>

#define NXD 768
#define NFD 768

typedef __attribute__((ext_vector_type(16))) __bf16 v16bf;
typedef __attribute__((ext_vector_type(8)))  float  v8f;

union Frag16 {
    v16bf  v;
    uint4  u[2];
    __bf16 h[16];
};

// ---------------------------------------------------------------------------
// K0: wt[f][d] = weight[d][f]  (one-time 2.25MB transpose so buildWt and the
//     GEMM B-path are both K-contiguous)
// ---------------------------------------------------------------------------
__global__ __launch_bounds__(256) void transposeW_kernel(
    const float* __restrict__ w, float* __restrict__ wt)
{
    const int idx = blockIdx.x * 256 + threadIdx.x;   // f*768 + d, d fast
    const int f = idx / NXD;
    const int d = idx - f * NXD;
    wt[idx] = w[(size_t)d * NFD + f];
}

// ---------------------------------------------------------------------------
// K1: wu[b,o] = sum_i h[b,i] * linW[o,i]   (both LoRA branches)
// ---------------------------------------------------------------------------
__global__ __launch_bounds__(256) void wu_kernel(
    const float* __restrict__ user_h, const float* __restrict__ item_h,
    const float* __restrict__ lora_U_w, const float* __restrict__ lora_I_w,
    float* __restrict__ wuU, float* __restrict__ wuI)
{
    __shared__ float sh[NXD];
    const int b  = blockIdx.x;
    const int br = blockIdx.y;
    const float* h  = (br ? item_h   : user_h)  + b * NXD;
    const float* lw =  br ? lora_I_w : lora_U_w;
    float*       wo = (br ? wuI      : wuU)     + b * NXD;

    for (int i = threadIdx.x; i < NXD; i += blockDim.x) sh[i] = h[i];
    __syncthreads();

    for (int o = threadIdx.x; o < NXD; o += blockDim.x) {
        const float* lr = lw + (size_t)o * NXD;
        float s = 0.f;
        #pragma unroll 4
        for (int i = 0; i < NXD; ++i) s = fmaf(lr[i], sh[i], s);
        wo[o] = s;
    }
}

// ---------------------------------------------------------------------------
// K2: Wt[b][f][d] = wt[f][d] + wuU[b,q*32+l]*VU[n,m] + wuI[b,q*32+l]*VI[n,m]
//     (d = q*32+n, f = l*24+m), split into bf16 hi + bf16 lo, K-contiguous.
// ---------------------------------------------------------------------------
__global__ __launch_bounds__(256) void buildWt_kernel(
    const float* __restrict__ wt,
    const float* __restrict__ wuU, const float* __restrict__ VU,
    const float* __restrict__ wuI, const float* __restrict__ VI,
    __bf16* __restrict__ Wthi, __bf16* __restrict__ Wtlo)
{
    const size_t e0 = ((size_t)blockIdx.x * blockDim.x + threadIdx.x) * 4;
    const int    d0 = (int)(e0 % NXD);
    const size_t bf = e0 / NXD;
    const int    f  = (int)(bf % NFD);
    const int    b  = (int)(bf / NFD);

    const int q = d0 >> 5;                  // same q for d0..d0+3 (4 | 32)
    const int l = f / 24;
    const int m = f - l * 24;
    const float aU = wuU[b * NXD + q * 32 + l];
    const float aI = wuI[b * NXD + q * 32 + l];

    #pragma unroll
    for (int j = 0; j < 4; ++j) {
        const int d = d0 + j;
        const int n = d & 31;
        const float w = wt[(size_t)f * NXD + d]
                      + aU * VU[n * 24 + m]
                      + aI * VI[n * 24 + m];
        const __bf16 hi = (__bf16)w;
        const __bf16 lo = (__bf16)(w - (float)hi);
        const size_t o = ((size_t)b * NFD + f) * NXD + d;
        Wthi[o] = hi;
        Wtlo[o] = lo;
    }
}

// ---------------------------------------------------------------------------
// K3: out = x @ W_b + bias via split-bf16 WMMA.
//     Block tile 128x128, 8 wave32s, wave tile 32x64, BK=32.
//     Double-buffered LDS; B tiles filled with GLOBAL_LOAD_ASYNC_TO_LDS_B128.
// ---------------------------------------------------------------------------
__global__ __launch_bounds__(256) void gemm_kernel(
    const float*  __restrict__ x,
    const __bf16* __restrict__ Wthi,
    const __bf16* __restrict__ Wtlo,
    const float*  __restrict__ bias,
    float*        __restrict__ out)
{
    __shared__ __bf16 Ahi[2][128][32];
    __shared__ __bf16 Alo[2][128][32];
    __shared__ __bf16 Bth[2][128][32];   // [n][k]
    __shared__ __bf16 Btl[2][128][32];

    const int colBase = blockIdx.x * 128;
    const int rowBase = blockIdx.y * 128;
    const int b       = rowBase >> 9;    // 512 rows per batch

    const int lane  = threadIdx.x & 31;
    const int wid   = threadIdx.x >> 5;
    const int mBase = (wid >> 1) * 32;
    const int cBase = (wid & 1) * 64;
    const int half  = lane >> 4;
    const int l16   = lane & 15;

    const v8f vzero = {0.f, 0.f, 0.f, 0.f, 0.f, 0.f, 0.f, 0.f};
    v8f acc[2][4];
    #pragma unroll
    for (int i = 0; i < 2; ++i)
        #pragma unroll
        for (int j = 0; j < 4; ++j) acc[i][j] = vzero;

    // A staging: thread -> (row, 16-wide K chunk)
    const int ar  = threadIdx.x >> 1;
    const int ac0 = (threadIdx.x & 1) * 16;
    const float* xrow = x + (size_t)(rowBase + ar) * NXD + ac0;

    // B staging: thread -> (hi/lo select, f-row); 4 async b128 per thread
    const int bsel = threadIdx.x >> 7;        // 0 = hi, 1 = lo
    const int brow = threadIdx.x & 127;
    const __bf16* gB = (bsel ? Wtlo : Wthi)
                     + ((size_t)(b * NFD + colBase + brow)) * NXD;

    float4 areg[4];

    auto loadA = [&](int kt) {
        #pragma unroll
        for (int j = 0; j < 4; ++j) areg[j] = *(const float4*)(xrow + kt + j * 4);
    };
    auto stageA = [&](int p) {
        Frag16 ph, pl;
        #pragma unroll
        for (int j = 0; j < 4; ++j) {
            const float vv[4] = {areg[j].x, areg[j].y, areg[j].z, areg[j].w};
            #pragma unroll
            for (int u = 0; u < 4; ++u) {
                const __bf16 hi = (__bf16)vv[u];
                ph.h[j * 4 + u] = hi;
                pl.h[j * 4 + u] = (__bf16)(vv[u] - (float)hi);
            }
        }
        *(uint4*)&Ahi[p][ar][ac0]     = ph.u[0];
        *(uint4*)&Ahi[p][ar][ac0 + 8] = ph.u[1];
        *(uint4*)&Alo[p][ar][ac0]     = pl.u[0];
        *(uint4*)&Alo[p][ar][ac0 + 8] = pl.u[1];
    };
    auto stageB = [&](int kt, int p) {
        __bf16* lb = bsel ? &Btl[p][brow][0] : &Bth[p][brow][0];
        const uint32_t lo = (uint32_t)(uintptr_t)lb;   // LDS aperture: addr[31:0]
        const __bf16* g = gB + kt;
        #pragma unroll
        for (int j = 0; j < 4; ++j)
            asm volatile("global_load_async_to_lds_b128 %0, %1, off"
                         :: "v"(lo + j * 16),
                            "v"((uint64_t)(uintptr_t)(g + j * 8))
                         : "memory");
    };

    // ---- prologue: stage tile 0 into buffer 0
    loadA(0);
    stageB(0, 0);
    stageA(0);
    asm volatile("s_wait_asynccnt 0x0" ::: "memory");
    __syncthreads();

    int p = 0;
    for (int it = 0; it < 24; ++it) {
        const bool hasNext = (it < 23);
        if (hasNext) {
            loadA((it + 1) * 32);        // global f32 loads, hidden behind WMMA
            stageB((it + 1) * 32, p ^ 1);// async global->LDS, hidden behind WMMA
        }

        #pragma unroll
        for (int wr = 0; wr < 2; ++wr) {
            const int m = mBase + wr * 16 + l16;
            Frag16 ah, al;
            ah.u[0] = *(const uint4*)(&Ahi[p][m][half * 8]);
            ah.u[1] = *(const uint4*)(&Ahi[p][m][16 + half * 8]);
            al.u[0] = *(const uint4*)(&Alo[p][m][half * 8]);
            al.u[1] = *(const uint4*)(&Alo[p][m][16 + half * 8]);
            #pragma unroll
            for (int c = 0; c < 4; ++c) {
                const int n = cBase + c * 16 + l16;
                Frag16 bh, bl;
                bh.u[0] = *(const uint4*)(&Bth[p][n][half * 8]);
                bh.u[1] = *(const uint4*)(&Bth[p][n][16 + half * 8]);
                bl.u[0] = *(const uint4*)(&Btl[p][n][half * 8]);
                bl.u[1] = *(const uint4*)(&Btl[p][n][16 + half * 8]);
                acc[wr][c] = __builtin_amdgcn_wmma_f32_16x16x32_bf16(
                    false, ah.v, false, bh.v, (short)0, acc[wr][c], false, false);
                acc[wr][c] = __builtin_amdgcn_wmma_f32_16x16x32_bf16(
                    false, ah.v, false, bl.v, (short)0, acc[wr][c], false, false);
                acc[wr][c] = __builtin_amdgcn_wmma_f32_16x16x32_bf16(
                    false, al.v, false, bh.v, (short)0, acc[wr][c], false, false);
            }
        }

        if (hasNext) {
            stageA(p ^ 1);
            asm volatile("s_wait_asynccnt 0x0" ::: "memory");
            __syncthreads();
        }
        p ^= 1;
    }

    // ---- epilogue: C/D layout -> global, + bias
    #pragma unroll
    for (int wr = 0; wr < 2; ++wr) {
        const int r0 = rowBase + mBase + wr * 16 + half * 8;
        #pragma unroll
        for (int c = 0; c < 4; ++c) {
            const int col = colBase + cBase + c * 16 + l16;
            const float bv = bias[col];
            #pragma unroll
            for (int v = 0; v < 8; ++v) {
                out[(size_t)(r0 + v) * NFD + col] = acc[wr][c][v] + bv;
            }
        }
    }
}

// ---------------------------------------------------------------------------
extern "C" void kernel_launch(void* const* d_in, const int* in_sizes, int n_in,
                              void* d_out, int out_size, void* d_ws, size_t ws_size,
                              hipStream_t stream) {
    const float* x      = (const float*)d_in[0];
    const float* user_h = (const float*)d_in[1];
    const float* item_h = (const float*)d_in[2];
    const float* weight = (const float*)d_in[3];
    const float* bias   = (const float*)d_in[4];
    const float* loraUw = (const float*)d_in[5];
    const float* VU     = (const float*)d_in[6];
    const float* loraIw = (const float*)d_in[7];
    const float* VI     = (const float*)d_in[8];
    float* out = (float*)d_out;

    char* ws = (char*)d_ws;
    float*  wuU  = (float*)ws;                               //    98304 B
    float*  wuI  = (float*)(ws + 98304);                     //    98304 B
    float*  wt   = (float*)(ws + 196608);                    //  2359296 B
    __bf16* Wthi = (__bf16*)(ws + 2555904);                  // 37748736 B
    __bf16* Wtlo = (__bf16*)(ws + 40304640ULL);              // 37748736 B

    transposeW_kernel<<<dim3(2304), 256, 0, stream>>>(weight, wt);
    wu_kernel<<<dim3(32, 2), 256, 0, stream>>>(user_h, item_h, loraUw, loraIw, wuU, wuI);
    buildWt_kernel<<<dim3(18432), 256, 0, stream>>>(wt, wuU, VU, wuI, VI, Wthi, Wtlo);
    gemm_kernel<<<dim3(6, 128), 256, 0, stream>>>(x, Wthi, Wtlo, bias, out);
}
